// VDTBARPolicy_41180146434496
// MI455X (gfx1250) — matrix operand order repaired and checked by reading, not compiled, not run on hardware
//
#include <hip/hip_runtime.h>

#define D_  1024
#define T_  2048
#define B_  2
#define H_  16
#define HD_ 64
#define L_  8
#define I_  4096
#define M_  (B_*T_)          // 4096 rows
#define MASKB (-10000.0f)

typedef __attribute__((ext_vector_type(16))) __bf16 v16bf;
typedef __attribute__((ext_vector_type(8)))  float  v8f;
typedef __attribute__((ext_vector_type(4)))  unsigned int v4u;
typedef __attribute__((ext_vector_type(8)))  unsigned int v8u;
typedef __attribute__((ext_vector_type(2)))  unsigned int v2u;
typedef __attribute__((ext_vector_type(4)))  float  fv4;

union Frag  { v16bf v; v4u u[2]; __bf16 e[16]; };
union Pack8 { v4u u; __bf16 e[8]; };
union Pack4 { v2u u; __bf16 e[4]; };

__device__ __forceinline__ v8f wmma_bf16(v16bf a, v16bf b, v8f c) {
#if defined(__HIP_DEVICE_COMPILE__)
  // D = A x B + C, 16x16x32 bf16 -> f32
  return __builtin_amdgcn_wmma_f32_16x16x32_bf16(false, a, false, b, (short)0, c, false, false);
#else
  (void)a; (void)b; return c;
#endif
}

__device__ __forceinline__ void wait_tensorcnt0() {
#if defined(__HIP_DEVICE_COMPILE__)
#if __has_builtin(__builtin_amdgcn_s_wait_tensorcnt)
  __builtin_amdgcn_s_wait_tensorcnt(0);
#else
  asm volatile("s_wait_tensorcnt 0x0" ::: "memory");
#endif
#endif
}

// ---------------- LayerNorm: fp32 in -> bf16 (or fp32) out ----------------
template<bool OUTB>
__global__ __launch_bounds__(256)
void ln_kernel(const float* __restrict__ x, const float* __restrict__ w,
               const float* __restrict__ b, void* __restrict__ out) {
  const int row = blockIdx.x;
  const int tid = threadIdx.x;
  const int lane = tid & 31, wid = tid >> 5;
  const fv4 v = *reinterpret_cast<const fv4*>(x + (size_t)row * D_ + tid * 4);
  float s1 = v[0] + v[1] + v[2] + v[3];
  float s2 = v[0]*v[0] + v[1]*v[1] + v[2]*v[2] + v[3]*v[3];
  #pragma unroll
  for (int o = 16; o >= 1; o >>= 1) { s1 += __shfl_xor(s1, o, 32); s2 += __shfl_xor(s2, o, 32); }
  __shared__ float r1[8], r2[8];
  if (lane == 0) { r1[wid] = s1; r2[wid] = s2; }
  __syncthreads();
  float t1 = 0.f, t2 = 0.f;
  #pragma unroll
  for (int i = 0; i < 8; ++i) { t1 += r1[i]; t2 += r2[i]; }
  const float mean = t1 * (1.0f / D_);
  const float var  = t2 * (1.0f / D_) - mean * mean;
  const float rsq  = rsqrtf(var + 1e-5f);
  const int c0 = tid * 4;
  float o4[4];
  #pragma unroll
  for (int j = 0; j < 4; ++j) o4[j] = (v[j] - mean) * rsq * w[c0 + j] + b[c0 + j];
  if (OUTB) {
    Pack4 p;
    #pragma unroll
    for (int j = 0; j < 4; ++j) p.e[j] = (__bf16)o4[j];
    *reinterpret_cast<v2u*>(((__bf16*)out) + (size_t)row * D_ + c0) = p.u;
  } else {
    fv4 ov = { o4[0], o4[1], o4[2], o4[3] };
    *reinterpret_cast<fv4*>(((float*)out) + (size_t)row * D_ + c0) = ov;
  }
}

// ---------------- GEMM: C[M,N] = A(bf16)[M,K] x W(f32->bf16)[K,N] + bias ----------------
// A tile (128x32 bf16) is brought into LDS by the Tensor Data Mover (tensor_load_to_lds),
// issued by wave 0 and synchronized with s_wait_tensorcnt + workgroup barrier.
// EPI 0: qkv epilogue (q,k row-major bf16; v scattered transposed into vT[b,h,hd,t])
// EPI 1: relu -> bf16
// EPI 2: residual(f32) + val -> f32
template<int EPI>
__global__ __launch_bounds__(256)
void gemm_kernel(const __bf16* __restrict__ A, const float* __restrict__ W,
                 const float* __restrict__ bias, const float* hres,
                 void* out, __bf16* vT, int N, int K) {
  __shared__ __bf16 As[128 * 32];
  __shared__ __bf16 Bs[128 * 32];
  const int tid  = threadIdx.x;
  const int lane = tid & 31, wid = tid >> 5;
  const int fr = lane & 15, hi = lane >> 4;
  const int blockN = blockIdx.x * 128, blockM = blockIdx.y * 128;
  const int waveM = wid >> 2, waveN = wid & 3;      // 2 x 4 wave grid -> 64x32 per wave

  v8f acc[4][2];
  #pragma unroll
  for (int i = 0; i < 4; ++i)
    #pragma unroll
    for (int j = 0; j < 2; ++j)
      #pragma unroll
      for (int r = 0; r < 8; ++r) acc[i][j][r] = 0.0f;

  const int bK = tid & 31, bC = (tid >> 5) * 16;

#if defined(__HIP_DEVICE_COMPILE__)
  // ---- Tensor DMA descriptor (D#) for the A tile: 2D tensor, bf16 ----
  const unsigned long long gArowBase =
      (unsigned long long)(const void*)(A + (size_t)blockM * K);
  const unsigned ldsOffA =
      (unsigned)(__attribute__((address_space(3))) char*)(void*)&As[0];
#endif

  for (int kk = 0; kk < K; kk += 32) {
    __syncthreads();

#if defined(__HIP_DEVICE_COMPILE__)
    if (wid == 0) {
      const unsigned long long gaddr = gArowBase + (unsigned long long)kk * 2ull;
      v4u g0;
      g0[0] = 1u;                                             // count=1, user descriptor
      g0[1] = ldsOffA;                                        // lds_addr (bytes)
      g0[2] = (unsigned)gaddr;                                // global_addr[31:0]
      g0[3] = (unsigned)((gaddr >> 32) & 0x01ffffffull)       // global_addr[56:32]
              | 0x80000000u;                                  // type=2 ("image")
      v8u g1;
      g1[0] = (1u << 16);                                     // data_size=2B, no multicast
      g1[1] = ((unsigned)K & 0xffffu) << 16;                  // tensor_dim0 lo16
      g1[2] = (((unsigned)K >> 16) & 0xffffu)                 // tensor_dim0 hi16
              | ((unsigned)M_ << 16);                         // tensor_dim1 lo16 (4096)
      g1[3] = (((unsigned)M_ >> 16) & 0xffffu)                // tensor_dim1 hi16
              | (32u << 16);                                  // tile_dim0 = 32
      g1[4] = 128u;                                           // tile_dim1 = 128, tile_dim2=0
      g1[5] = (unsigned)K;                                    // tensor_dim0_stride lo32
      g1[6] = 0u;                                             // stride hi + dim1_stride lo
      g1[7] = 0u;
      asm volatile("tensor_load_to_lds %0, %1"
                   :: "s"(g0), "s"(g1) : "memory");
    }
#endif

    // B tile: transpose + fp32->bf16: Bs[n][k] = W[kk+k][blockN+n]
    const float* wg = W + (size_t)(kk + bK) * N + blockN + bC;
    if (kk + 32 < K) __builtin_prefetch(wg + (size_t)32 * N, 0, 1);
    float wv[16];
    #pragma unroll
    for (int q4 = 0; q4 < 4; ++q4) {
      fv4 t = *reinterpret_cast<const fv4*>(wg + q4 * 4);
      wv[q4*4+0]=t[0]; wv[q4*4+1]=t[1]; wv[q4*4+2]=t[2]; wv[q4*4+3]=t[3];
    }
    #pragma unroll
    for (int j = 0; j < 16; ++j) Bs[(bC + j) * 32 + bK] = (__bf16)wv[j];

#if defined(__HIP_DEVICE_COMPILE__)
    if (wid == 0) wait_tensorcnt0();
#endif
    __syncthreads();

    // fragments + WMMA
    v16bf af[4], bf2[2];
    const int aOff = hi ? 8 : 0;    // A layout: K 0..7&16..23 / 8..15&24..31
    #pragma unroll
    for (int mt = 0; mt < 4; ++mt) {
      const __bf16* p = &As[(waveM * 64 + mt * 16 + fr) * 32];
      Frag f;
      f.u[0] = *reinterpret_cast<const v4u*>(p + aOff);
      f.u[1] = *reinterpret_cast<const v4u*>(p + aOff + 16);
      af[mt] = f.v;
    }
    const int bOff = hi ? 8 : 0;    // B layout: K 0..7&16..23 / 8..15&24..31
    #pragma unroll
    for (int nt = 0; nt < 2; ++nt) {
      const __bf16* p = &Bs[(waveN * 32 + nt * 16 + fr) * 32];
      Frag f;
      f.u[0] = *reinterpret_cast<const v4u*>(p + bOff);
      f.u[1] = *reinterpret_cast<const v4u*>(p + bOff + 16);
      bf2[nt] = f.v;
    }
    #pragma unroll
    for (int mt = 0; mt < 4; ++mt)
      #pragma unroll
      for (int nt = 0; nt < 2; ++nt)
        acc[mt][nt] = wmma_bf16(af[mt], bf2[nt], acc[mt][nt]);
  }

  // epilogue: C layout: vgpr r -> M = r + 8*hi ; N = fr
  #pragma unroll
  for (int mt = 0; mt < 4; ++mt) {
    #pragma unroll
    for (int nt = 0; nt < 2; ++nt) {
      const int col = blockN + waveN * 32 + nt * 16 + fr;
      const float bv = bias[col];
      #pragma unroll
      for (int r = 0; r < 8; ++r) {
        const int row = blockM + waveM * 64 + mt * 16 + r + hi * 8;
        const float val = acc[mt][nt][r] + bv;
        if (EPI == 0) {
          if (col < 2 * D_) {
            ((__bf16*)out)[(size_t)row * N + col] = (__bf16)val;
          } else {
            const int dall = col - 2 * D_;
            const int hh = dall >> 6, dd = dall & 63;
            const int bidx = row >> 11, t = row & (T_ - 1);
            vT[((size_t)(bidx * H_ + hh) * HD_ + dd) * T_ + t] = (__bf16)val;
          }
        } else if (EPI == 1) {
          ((__bf16*)out)[(size_t)row * N + col] = (__bf16)fmaxf(val, 0.0f);
        } else {
          ((float*)out)[(size_t)row * N + col] = hres[(size_t)row * N + col] + val;
        }
      }
    }
  }
}

// ---------------- Flash attention (transposed-score formulation) ----------------
// S^T = K * Q^T (M=key, N=query);  O^T = V^T * P^T (M=hd, N=query)
__global__ __launch_bounds__(128)
void attn_kernel(const __bf16* __restrict__ qkv, const __bf16* __restrict__ vT,
                 const float* __restrict__ amask, __bf16* __restrict__ ctx) {
  const int tid = threadIdx.x, lane = tid & 31, wid = tid >> 5;
  const int fr = lane & 15, hi = lane >> 4;
  const int bh = blockIdx.y, b = bh >> 4, h = bh & 15;
  const int q0 = blockIdx.x * 64 + wid * 16;      // wave-private 16 query rows
  const size_t rstride = 3 * D_;
  const __bf16* qbase = qkv + (size_t)(b * T_) * rstride + h * HD_;
  const __bf16* kbase = qbase + D_;
  const __bf16* vtb   = vT + (size_t)bh * HD_ * T_;

  // Q^T B-fragments (N=query=fr; K=hd: half 0..7&16..23 / 8..15&24..31)
  v16bf qf[2];
  #pragma unroll
  for (int s = 0; s < 2; ++s) {
    const __bf16* p = qbase + (size_t)(q0 + fr) * rstride + s * 32 + hi * 8;
    Frag f;
    f.u[0] = *reinterpret_cast<const v4u*>(p);
    f.u[1] = *reinterpret_cast<const v4u*>(p + 16);
    qf[s] = f.v;
  }

  v8f accO[4];
  #pragma unroll
  for (int mt = 0; mt < 4; ++mt)
    #pragma unroll
    for (int r = 0; r < 8; ++r) accO[mt][r] = 0.0f;

  float m_run = -1e30f, l_run = 0.0f;
  const int myq = q0 + fr;
  const float scale = 0.125f;   // 1/sqrt(64)

  for (int k0 = 0; k0 < q0 + 16; k0 += 32) {
    float p0[8], p1[8];
    // --- two 16(key)x16(query) S^T tiles over hd=64 (two K-steps of 32) ---
    #pragma unroll
    for (int t = 0; t < 2; ++t) {
      v8f c;
      #pragma unroll
      for (int r = 0; r < 8; ++r) c[r] = 0.0f;
      #pragma unroll
      for (int s = 0; s < 2; ++s) {
        const __bf16* kp = kbase + (size_t)(k0 + t * 16 + fr) * rstride + s * 32 + hi * 8;
        Frag f;   // A-fragment: row=key, K=hd (A layout)
        f.u[0] = *reinterpret_cast<const v4u*>(kp);
        f.u[1] = *reinterpret_cast<const v4u*>(kp + 16);
        c = wmma_bf16(f.v, qf[s], c);
      }
      float* dst = t ? p1 : p0;
      #pragma unroll
      for (int r = 0; r < 8; ++r) {
        const int key = k0 + t * 16 + r + hi * 8;
        float sv = c[r] * scale;
        sv = (key > myq) ? MASKB : sv;                       // causal masked_fill
        sv += (1.0f - amask[b * T_ + key]) * MASKB;          // additive mask
        dst[r] = sv;
      }
    }
    // --- online softmax over keys (per query = fr; halves combined via xor16) ---
    float rm = p0[0];
    #pragma unroll
    for (int r = 1; r < 8; ++r) rm = fmaxf(rm, p0[r]);
    #pragma unroll
    for (int r = 0; r < 8; ++r) rm = fmaxf(rm, p1[r]);
    rm = fmaxf(rm, __shfl_xor(rm, 16, 32));
    const float m_new = fmaxf(m_run, rm);
    const float alpha = __expf(m_run - m_new);
    float rs = 0.0f;
    #pragma unroll
    for (int r = 0; r < 8; ++r) {
      p0[r] = __expf(p0[r] - m_new);
      p1[r] = __expf(p1[r] - m_new);
      rs += p0[r] + p1[r];
    }
    rs += __shfl_xor(rs, 16, 32);
    l_run = l_run * alpha + rs;
    m_run = m_new;
    #pragma unroll
    for (int mt = 0; mt < 4; ++mt)
      #pragma unroll
      for (int r = 0; r < 8; ++r) accO[mt][r] *= alpha;
    // --- pack P^T B-fragment: lane needs its query's keys 0..15 (lo) / 16..31 (hi) ---
    float x0[8], x1[8];
    #pragma unroll
    for (int r = 0; r < 8; ++r) { x0[r] = __shfl_xor(p0[r], 16, 32); x1[r] = __shfl_xor(p1[r], 16, 32); }
    Frag pf;
    #pragma unroll
    for (int r = 0; r < 8; ++r) {
      pf.e[r]     = (__bf16)(hi ? x1[r] : p0[r]);
      pf.e[8 + r] = (__bf16)(hi ? p1[r] : x0[r]);
    }
    // --- O^T += V^T x P^T : four 16(hd)x16(q) tiles, K=32 keys ---
    #pragma unroll
    for (int mt = 0; mt < 4; ++mt) {
      const __bf16* vp = vtb + (size_t)(mt * 16 + fr) * T_ + k0 + hi * 16;
      Frag f;   // A-fragment: row=hd, K=keys (contiguous thanks to vT layout)
      f.u[0] = *reinterpret_cast<const v4u*>(vp);
      f.u[1] = *reinterpret_cast<const v4u*>(vp + 8);
      accO[mt] = wmma_bf16(f.v, pf.v, accO[mt]);
    }
  }

  const float inv = 1.0f / l_run;
  __bf16* orow = ctx + (size_t)(b * T_ + myq) * D_ + h * HD_;
  #pragma unroll
  for (int mt = 0; mt < 4; ++mt) {
    Pack8 pk;
    #pragma unroll
    for (int r = 0; r < 8; ++r) pk.e[r] = (__bf16)(accO[mt][r] * inv);
    *reinterpret_cast<v4u*>(orow + mt * 16 + hi * 8) = pk.u;
  }
}

// ---------------- driver ----------------
extern "C" void kernel_launch(void* const* d_in, const int* in_sizes, int n_in,
                              void* d_out, int out_size, void* d_ws, size_t ws_size,
                              hipStream_t stream) {
  const float* embeds = (const float*)d_in[0];
  const float* amask  = (const float*)d_in[1];
  const float* ln1w = (const float*)d_in[2],  * ln1b = (const float*)d_in[3];
  const float* attw = (const float*)d_in[4],  * attb = (const float*)d_in[5];
  const float* prjw = (const float*)d_in[6],  * prjb = (const float*)d_in[7];
  const float* ln2w = (const float*)d_in[8],  * ln2b = (const float*)d_in[9];
  const float* fcw  = (const float*)d_in[10], * fcb  = (const float*)d_in[11];
  const float* fpw  = (const float*)d_in[12], * fpb  = (const float*)d_in[13];
  const float* lnfw = (const float*)d_in[14], * lnfb = (const float*)d_in[15];

  char* ws = (char*)d_ws;
  float*  h   = (float*)ws;                                   // 16 MB fp32 residual
  __bf16* x   = (__bf16*)(ws + ((size_t)16 << 20));           // 8 MB  (LN out / ctx)
  __bf16* qkv = (__bf16*)(ws + ((size_t)24 << 20));           // 24 MB (reused as ffn1, 32 MB)
  __bf16* vT  = (__bf16*)(ws + ((size_t)48 << 20));           // 8 MB
  __bf16* ffn = qkv;

  hipMemcpyAsync(h, embeds, (size_t)M_ * D_ * sizeof(float), hipMemcpyDeviceToDevice, stream);

  for (int i = 0; i < L_; ++i) {
    ln_kernel<true><<<M_, 256, 0, stream>>>(h, ln1w + (size_t)i * D_, ln1b + (size_t)i * D_, x);
    gemm_kernel<0><<<dim3(3 * D_ / 128, M_ / 128), 256, 0, stream>>>(
        x, attw + (size_t)i * D_ * 3 * D_, attb + (size_t)i * 3 * D_,
        nullptr, qkv, vT, 3 * D_, D_);
    attn_kernel<<<dim3(T_ / 64, B_ * H_), 128, 0, stream>>>(qkv, vT, amask, x); // ctx -> x
    gemm_kernel<2><<<dim3(D_ / 128, M_ / 128), 256, 0, stream>>>(
        x, prjw + (size_t)i * D_ * D_, prjb + (size_t)i * D_,
        h, h, nullptr, D_, D_);
    ln_kernel<true><<<M_, 256, 0, stream>>>(h, ln2w + (size_t)i * D_, ln2b + (size_t)i * D_, x);
    gemm_kernel<1><<<dim3(I_ / 128, M_ / 128), 256, 0, stream>>>(
        x, fcw + (size_t)i * D_ * I_, fcb + (size_t)i * I_,
        nullptr, ffn, nullptr, I_, D_);
    gemm_kernel<2><<<dim3(D_ / 128, M_ / 128), 256, 0, stream>>>(
        ffn, fpw + (size_t)i * I_ * D_, fpb + (size_t)i * D_,
        h, h, nullptr, D_, I_);
  }
  ln_kernel<false><<<M_, 256, 0, stream>>>(h, lnfw, lnfb, (float*)d_out);
}
